// LlamaMLP_HalfwayGIN_28853590294728
// MI455X (gfx1250) — matrix operand order repaired and checked by reading, not compiled, not run on hardware
//
#include <hip/hip_runtime.h>

typedef __attribute__((ext_vector_type(8)))  __bf16 bf16x8;
typedef __attribute__((ext_vector_type(16))) __bf16 bf16x16;
typedef __attribute__((ext_vector_type(8)))  float  v8f;
typedef __attribute__((ext_vector_type(4)))  unsigned int u32x4;
typedef __attribute__((ext_vector_type(8)))  unsigned int u32x8;

#define BB 2
#define SS 2048
#define DD 2048
#define HH 16
#define II 8192
#define HDD 512

#define TM 128
#define TN 128
#define KS 32

// ---------------- fragment loaders (contiguous k = 16*(lane/16) + s, same for A and B) ----------------

// A 16x32 bf16: lane<16 holds row M=lane, K=0..15 contiguous; lane>=16 holds
// row M=lane-16, K=16..31. LDS tile is row-major [rows][32].
__device__ inline bf16x16 load_a_frag(const __bf16* lds, int mBase, int lane) {
  int m   = mBase + (lane & 15);
  int klo = (lane >> 4) * 16;
  const __bf16* p = lds + m * KS + klo;
  bf16x8 lo = *(const bf16x8*)p;
  bf16x8 hi = *(const bf16x8*)(p + 8);
  return __builtin_shufflevector(lo, hi, 0,1,2,3,4,5,6,7,8,9,10,11,12,13,14,15);
}

// B 32x16 bf16: lane<16 holds col N=lane with K=0..15 contiguous; lane>=16 holds
// col N=lane-16 with K=16..31. LDS tile is N-major [cols][32] (K contiguous).
__device__ inline bf16x16 load_b_frag(const __bf16* lds, int nBase, int lane) {
  int n  = nBase + (lane & 15);
  int k0 = (lane >> 4) * 16;
  const __bf16* p = lds + n * KS + k0;
  bf16x8 lo = *(const bf16x8*)p;
  bf16x8 hi = *(const bf16x8*)(p + 8);
  return __builtin_shufflevector(lo, hi, 0,1,2,3,4,5,6,7,8,9,10,11,12,13,14,15);
}

__device__ inline v8f wmma_bf16(bf16x16 a, bf16x16 b, v8f c) {
  return __builtin_amdgcn_wmma_f32_16x16x32_bf16(false, a, false, b, (short)0, c,
                                                 false, false);
}

__device__ inline bf16x8 cvt8(const float* s) {
  float4 f0 = *(const float4*)(s);
  float4 f1 = *(const float4*)(s + 4);
  bf16x8 v;
  v[0] = (__bf16)f0.x; v[1] = (__bf16)f0.y; v[2] = (__bf16)f0.z; v[3] = (__bf16)f0.w;
  v[4] = (__bf16)f1.x; v[5] = (__bf16)f1.y; v[6] = (__bf16)f1.z; v[7] = (__bf16)f1.w;
  return v;
}

// ---------------- Tensor Data Mover: 2D tile DMA global -> LDS ----------------
// Builds a 2-group D# (tensors up to 2D need only groups 0/1; VADDR2/3 = NULL)
// and issues TENSOR_LOAD_TO_LDS. Tracked by TENSORcnt.
// Tile: rows x 32 bf16 elements, row pitch = pitch_elems (in elements).
__device__ inline void tdm_load_tile(__bf16* lds, const __bf16* gsrc,
                                     unsigned rows, unsigned pitch_elems) {
  unsigned long long ga = (unsigned long long)(uintptr_t)gsrc;
  unsigned la = (unsigned)(uintptr_t)lds;  // low 32 bits of generic = LDS byte addr
  u32x4 g0;
  g0[0] = 1u;                                   // count=1, is_restore=0, no gather
  g0[1] = la;                                   // lds_addr [63:32]
  g0[2] = (unsigned)ga;                         // global_addr [95:64]
  g0[3] = (unsigned)(ga >> 32) | 0x80000000u;   // global_addr[56:32] | type=2
  const unsigned TD = 1u << 20;  // large tensor dims: no OOB clipping
  u32x8 g1;
  g1[0] = 0x00010000u;                          // wg_mask=0, data_size=1 (2 bytes)
  g1[1] = (TD & 0xffffu) << 16;                 // tensor_dim0[15:0]
  g1[2] = (TD >> 16) | ((TD & 0xffffu) << 16);  // tensor_dim0[31:16] | tensor_dim1[15:0]
  g1[3] = (TD >> 16) | ((unsigned)KS << 16);    // tensor_dim1[31:16] | tile_dim0=32
  g1[4] = rows & 0xffffu;                       // tile_dim1 | tile_dim2=0
  g1[5] = pitch_elems;                          // tensor_dim0_stride[31:0]
  g1[6] = 0u;                                   // stride[47:32] | dim1_stride[15:0]
  g1[7] = 0u;
  asm volatile("tensor_load_to_lds %0, %1" :: "s"(g0), "s"(g1) : "memory");
}

__device__ inline void tdm_wait() { __builtin_amdgcn_s_wait_tensorcnt(0); }

// ---------------- pre-pass kernels ----------------

__global__ __launch_bounds__(256) void k_cvt_bf16(const float* __restrict__ in,
                                                  __bf16* __restrict__ out, int n4) {
  int i = blockIdx.x * blockDim.x + threadIdx.x;
  if (i < n4) {
    float4 f = *(const float4*)(in + (size_t)i * 4);
    __bf16 o[4] = {(__bf16)f.x, (__bf16)f.y, (__bf16)f.z, (__bf16)f.w};
    *(uint2*)(out + (size_t)i * 4) = *(uint2*)o;
  }
}

// in: [R][C] f32 -> out: [C][R] bf16
__global__ __launch_bounds__(256) void k_transpose_cvt(const float* __restrict__ in,
                                                       __bf16* __restrict__ out,
                                                       int R, int C) {
  __shared__ float t[32][33];
  int bx = blockIdx.x * 32;  // C
  int by = blockIdx.y * 32;  // R
  int tx = threadIdx.x & 31;
  int ty = threadIdx.x >> 5;  // 0..7
#pragma unroll
  for (int i = 0; i < 32; i += 8)
    t[ty + i][tx] = in[(size_t)(by + ty + i) * C + bx + tx];
  __syncthreads();
#pragma unroll
  for (int i = 0; i < 32; i += 8)
    out[(size_t)(bx + ty + i) * R + by + tx] = (__bf16)t[tx][ty + i];
}

// ---------------- GEMM 1: fused gate/up + SiLU, writes h_heads^T ----------------
// xb   : [4096][2048] bf16 ; WgT/WuT: [8192][2048] bf16 (N-major)
// hhT  : [32][512][2048] bf16  (bh, hd, s)
// Waves 4(M)x2(N); wave tile 32x64, dual accumulators: 16 wmma / k-step.
__global__ __launch_bounds__(256) void k_gemm_gateup(const __bf16* __restrict__ xb,
                                                     const __bf16* __restrict__ WgT,
                                                     const __bf16* __restrict__ WuT,
                                                     __bf16* __restrict__ hhT) {
  __shared__ alignas(16) __bf16 lA[2][TM * KS];
  __shared__ alignas(16) __bf16 lG[2][TN * KS];
  __shared__ alignas(16) __bf16 lU[2][TN * KS];

  const int tid  = threadIdx.x;
  const int lane = tid & 31;
  const int wave = tid >> 5;
  const int wm = (wave >> 1) * 32;  // 0,32,64,96
  const int wn = (wave & 1) * 64;   // 0,64
  const int bm = blockIdx.y * TM;   // bs
  const int bn = blockIdx.x * TN;   // i

  const __bf16* Abase = xb  + (size_t)bm * DD;
  const __bf16* Gbase = WgT + (size_t)bn * DD;
  const __bf16* Ubase = WuT + (size_t)bn * DD;

  const v8f vz = {0.f, 0.f, 0.f, 0.f, 0.f, 0.f, 0.f, 0.f};
  v8f accg[2][4], accu[2][4];
#pragma unroll
  for (int i = 0; i < 2; i++)
#pragma unroll
    for (int j = 0; j < 4; j++) { accg[i][j] = vz; accu[i][j] = vz; }

  if (wave == 0) {
    tdm_load_tile(&lA[0][0], Abase, TM, DD);
    tdm_load_tile(&lG[0][0], Gbase, TN, DD);
    tdm_load_tile(&lU[0][0], Ubase, TN, DD);
    tdm_wait();
  }
  __syncthreads();

  const int nk = DD / KS;
  for (int i = 0; i < nk; ++i) {
    const int cur = i & 1, nxt = cur ^ 1;
    if (wave == 0 && i + 1 < nk) {
      const int k = (i + 1) * KS;
      tdm_load_tile(&lA[nxt][0], Abase + k, TM, DD);
      tdm_load_tile(&lG[nxt][0], Gbase + k, TN, DD);
      tdm_load_tile(&lU[nxt][0], Ubase + k, TN, DD);
    }
    bf16x16 a0 = load_a_frag(lA[cur], wm + 0,  lane);
    bf16x16 a1 = load_a_frag(lA[cur], wm + 16, lane);
#pragma unroll
    for (int tn = 0; tn < 4; ++tn) {
      bf16x16 g = load_b_frag(lG[cur], wn + tn * 16, lane);
      bf16x16 u = load_b_frag(lU[cur], wn + tn * 16, lane);
      accg[0][tn] = wmma_bf16(a0, g, accg[0][tn]);
      accg[1][tn] = wmma_bf16(a1, g, accg[1][tn]);
      accu[0][tn] = wmma_bf16(a0, u, accu[0][tn]);
      accu[1][tn] = wmma_bf16(a1, u, accu[1][tn]);
    }
    if (wave == 0) tdm_wait();
    __syncthreads();
  }

#pragma unroll
  for (int tm = 0; tm < 2; tm++)
#pragma unroll
    for (int tn = 0; tn < 4; tn++) {
      int row0 = bm + wm + tm * 16 + (lane >> 4) * 8;  // bs, 8 consecutive rows
      int col  = bn + wn + tn * 16 + (lane & 15);      // i
      int h = col >> 9, hd = col & 511;
      int b = row0 >> 11, s0 = row0 & 2047;
      bf16x8 o;
#pragma unroll
      for (int r = 0; r < 8; r++) {
        float g = accg[tm][tn][r];
        float u = accu[tm][tn][r];
        float sg = g / (1.0f + __expf(-g));  // SiLU
        o[r] = (__bf16)(sg * u);
      }
      *(bf16x8*)(hhT + ((size_t)((b * HH + h) * HDD + hd)) * SS + s0) = o;
    }
}

// ---------------- GEMM 2: agg = adj @ h_heads, GIN combine ----------------
// adj: [32][2048][2048] f32 (VALU-converted while staging; TDM can't convert)
// hhT: [32][512][2048] bf16 (B operand via TDM) ; hcomb: [4096][8192] bf16
__global__ __launch_bounds__(256) void k_gemm_adj(const float* __restrict__ adj,
                                                  const __bf16* __restrict__ hhT,
                                                  const float* __restrict__ eps,
                                                  const float* __restrict__ alpha,
                                                  __bf16* __restrict__ hcomb) {
  __shared__ alignas(16) __bf16 lA[2][TM * KS];
  __shared__ alignas(16) __bf16 lB[2][TN * KS];

  const int tid  = threadIdx.x;
  const int lane = tid & 31;
  const int wave = tid >> 5;
  const int wm = (wave >> 1) * 32;
  const int wn = (wave & 1) * 64;
  const int bm = blockIdx.y * TM;  // s
  const int bn = blockIdx.x * TN;  // hd
  const int bh = blockIdx.z;
  const int b = bh >> 4, h = bh & 15;
  const float e  = 1.0f + eps[h];
  const float al = alpha[h];

  const float*  adjp = adj + (size_t)bh * SS * SS + (size_t)bm * SS;
  const __bf16* hp   = hhT + (size_t)bh * HDD * SS;
  const __bf16* Bbase = hp + (size_t)bn * SS;

  const v8f vz = {0.f, 0.f, 0.f, 0.f, 0.f, 0.f, 0.f, 0.f};
  v8f acc[2][4];
#pragma unroll
  for (int i = 0; i < 2; i++)
#pragma unroll
    for (int j = 0; j < 4; j++) acc[i][j] = vz;

  const int arow = tid >> 1, ahalf = (tid & 1) * 16;  // A: 128 rows x 2 halves

  // prologue: stage buffer 0
  {
    const float* s = adjp + (size_t)arow * SS + ahalf;
    *(bf16x8*)(&lA[0][arow * KS + ahalf])     = cvt8(s);
    *(bf16x8*)(&lA[0][arow * KS + ahalf + 8]) = cvt8(s + 8);
    if (wave == 0) {
      tdm_load_tile(&lB[0][0], Bbase, TN, SS);
      tdm_wait();
    }
  }
  __syncthreads();

  const int nk = SS / KS;
  for (int i = 0; i < nk; ++i) {
    const int cur = i & 1, nxt = cur ^ 1;
    if (i + 1 < nk) {
      const int k = (i + 1) * KS;
      const float* s = adjp + (size_t)arow * SS + k + ahalf;
      *(bf16x8*)(&lA[nxt][arow * KS + ahalf])     = cvt8(s);
      *(bf16x8*)(&lA[nxt][arow * KS + ahalf + 8]) = cvt8(s + 8);
      if (wave == 0) tdm_load_tile(&lB[nxt][0], Bbase + k, TN, SS);
    }
    bf16x16 a0 = load_a_frag(lA[cur], wm + 0,  lane);
    bf16x16 a1 = load_a_frag(lA[cur], wm + 16, lane);
#pragma unroll
    for (int tn = 0; tn < 4; ++tn) {
      bf16x16 bb = load_b_frag(lB[cur], wn + tn * 16, lane);
      acc[0][tn] = wmma_bf16(a0, bb, acc[0][tn]);
      acc[1][tn] = wmma_bf16(a1, bb, acc[1][tn]);
    }
    if (wave == 0) tdm_wait();
    __syncthreads();
  }

#pragma unroll
  for (int tm = 0; tm < 2; tm++)
#pragma unroll
    for (int tn = 0; tn < 4; tn++) {
      int row0 = bm + wm + tm * 16 + (lane >> 4) * 8;  // s
      int col  = bn + wn + tn * 16 + (lane & 15);      // hd
      bf16x8 hh8 = *(const bf16x8*)(hp + (size_t)col * SS + row0);
#pragma unroll
      for (int r = 0; r < 8; r++) {
        float v = e * (float)hh8[r] + al * acc[tm][tn][r];
        hcomb[((size_t)(b * SS + row0 + r)) * II + h * HDD + col] = (__bf16)v;
      }
    }
}

// ---------------- GEMM 3: out = h_comb @ Wd ----------------
// hcomb: [4096][8192] bf16 ; WdT: [2048][8192] bf16 (N-major) ; out f32
__global__ __launch_bounds__(256) void k_gemm_down(const __bf16* __restrict__ hcomb,
                                                   const __bf16* __restrict__ WdT,
                                                   float* __restrict__ out) {
  __shared__ alignas(16) __bf16 lA[2][TM * KS];
  __shared__ alignas(16) __bf16 lB[2][TN * KS];

  const int tid  = threadIdx.x;
  const int lane = tid & 31;
  const int wave = tid >> 5;
  const int wm = (wave >> 1) * 32;
  const int wn = (wave & 1) * 64;
  const int bm = blockIdx.y * TM;  // bs
  const int bn = blockIdx.x * TN;  // d

  const __bf16* Abase = hcomb + (size_t)bm * II;
  const __bf16* Bbase = WdT   + (size_t)bn * II;

  const v8f vz = {0.f, 0.f, 0.f, 0.f, 0.f, 0.f, 0.f, 0.f};
  v8f acc[2][4];
#pragma unroll
  for (int i = 0; i < 2; i++)
#pragma unroll
    for (int j = 0; j < 4; j++) acc[i][j] = vz;

  if (wave == 0) {
    tdm_load_tile(&lA[0][0], Abase, TM, II);
    tdm_load_tile(&lB[0][0], Bbase, TN, II);
    tdm_wait();
  }
  __syncthreads();

  const int nk = II / KS;
  for (int i = 0; i < nk; ++i) {
    const int cur = i & 1, nxt = cur ^ 1;
    if (wave == 0 && i + 1 < nk) {
      const int k = (i + 1) * KS;
      tdm_load_tile(&lA[nxt][0], Abase + k, TM, II);
      tdm_load_tile(&lB[nxt][0], Bbase + k, TN, II);
    }
    bf16x16 a0 = load_a_frag(lA[cur], wm + 0,  lane);
    bf16x16 a1 = load_a_frag(lA[cur], wm + 16, lane);
#pragma unroll
    for (int tn = 0; tn < 4; ++tn) {
      bf16x16 bb = load_b_frag(lB[cur], wn + tn * 16, lane);
      acc[0][tn] = wmma_bf16(a0, bb, acc[0][tn]);
      acc[1][tn] = wmma_bf16(a1, bb, acc[1][tn]);
    }
    if (wave == 0) tdm_wait();
    __syncthreads();
  }

#pragma unroll
  for (int tm = 0; tm < 2; tm++)
#pragma unroll
    for (int tn = 0; tn < 4; tn++) {
      int row0 = bm + wm + tm * 16 + (lane >> 4) * 8;
      int col  = bn + wn + tn * 16 + (lane & 15);
#pragma unroll
      for (int r = 0; r < 8; r++)
        out[(size_t)(row0 + r) * DD + col] = acc[tm][tn][r];
    }
}

// ---------------- host launcher ----------------

extern "C" void kernel_launch(void* const* d_in, const int* in_sizes, int n_in,
                              void* d_out, int out_size, void* d_ws, size_t ws_size,
                              hipStream_t stream) {
  const float* x    = (const float*)d_in[0];  // [2][2048][2048]
  const float* adj  = (const float*)d_in[1];  // [2][16][2048][2048]
  const float* Wg   = (const float*)d_in[2];  // [2048][8192]
  const float* Wu   = (const float*)d_in[3];  // [2048][8192]
  const float* Wd   = (const float*)d_in[4];  // [8192][2048]
  const float* eps  = (const float*)d_in[5];  // [16]
  const float* alp  = (const float*)d_in[6];  // [16]
  float* out = (float*)d_out;

  char* ws = (char*)d_ws;
  __bf16* xb    = (__bf16*)(ws);                        // 16 MiB
  __bf16* WgT   = (__bf16*)(ws + 16777216ull);          // 32 MiB
  __bf16* WuT   = (__bf16*)(ws + 50331648ull);          // 32 MiB
  __bf16* WdT   = (__bf16*)(ws + 83886080ull);          // 32 MiB
  __bf16* hhT   = (__bf16*)(ws + 117440512ull);         // 64 MiB
  __bf16* hcomb = (__bf16*)(ws + 184549376ull);         // 64 MiB

  {
    int n4 = (BB * SS * DD) / 4;
    k_cvt_bf16<<<(n4 + 255) / 256, 256, 0, stream>>>(x, xb, n4);
  }
  k_transpose_cvt<<<dim3(II / 32, DD / 32), 256, 0, stream>>>(Wg, WgT, DD, II);
  k_transpose_cvt<<<dim3(II / 32, DD / 32), 256, 0, stream>>>(Wu, WuT, DD, II);
  k_transpose_cvt<<<dim3(DD / 32, II / 32), 256, 0, stream>>>(Wd, WdT, II, DD);

  k_gemm_gateup<<<dim3(II / TN, (BB * SS) / TM), 256, 0, stream>>>(xb, WgT, WuT, hhT);
  k_gemm_adj<<<dim3(HDD / TN, SS / TM, BB * HH), 256, 0, stream>>>(adj, hhT, eps, alp,
                                                                   hcomb);
  k_gemm_down<<<dim3(DD / TN, (BB * SS) / TM), 256, 0, stream>>>(hcomb, WdT, out);
}
